// VectorQuantizer_85023172591763
// MI455X (gfx1250) — matrix operand order repaired and checked by reading, not compiled, not run on hardware
//
#include <hip/hip_runtime.h>
#include <hip/hip_bf16.h>
#include <math.h>

// ---------------------------------------------------------------------------
// VQ-VAE vector quantizer for MI455X (gfx1250, wave32, WMMA).
//   N = 16*1024 = 16384 tokens, D = 256, K = 4096 codes.
//   dist = 2 - 2*cos  =>  softmax(-dist/0.1) == softmax(20*cos) (shift cancels)
//   argmin(dist) == argmax(cos), ties -> lowest index (matches jnp.argmin).
// ---------------------------------------------------------------------------

typedef __attribute__((ext_vector_type(16))) _Float16 v16h;
typedef __attribute__((ext_vector_type(8)))  _Float16 v8h;
typedef __attribute__((ext_vector_type(8)))  float    v8f;

#define N_ROWS   16384
#define D_DIM    256
#define K_CB     4096
#define ROWS_PER_BLK 64         // 4 row-tiles of 16
#define LOGIT_SCALE  20.0f      // 2 / TEMP

// ---------------------------------------------------------------------------
// Row-wise L2 normalize: one block per row, 256 threads == D.
// Emits f32 (for gather / loss) and f16 (for WMMA).
// ---------------------------------------------------------------------------
__global__ __launch_bounds__(256)
void k_normalize(const float* __restrict__ src,
                 float* __restrict__ dst32,
                 _Float16* __restrict__ dst16) {
    __shared__ float red[256];
    const int t = threadIdx.x;
    const size_t base = (size_t)blockIdx.x * D_DIM;
    float v = src[base + t];
    red[t] = v * v;
    __syncthreads();
    #pragma unroll
    for (int s = 128; s > 0; s >>= 1) {
        if (t < s) red[t] += red[t + s];
        __syncthreads();
    }
    const float inv = 1.0f / fmaxf(sqrtf(red[0]), 1e-12f);
    v *= inv;
    dst32[base + t] = v;
    dst16[base + t] = (_Float16)v;
}

// ---------------------------------------------------------------------------
// Fused GEMM (WMMA f16) + softmax + argmax per 64-row block.
//   grid = 256 blocks, block = 256 threads (8 waves).
//   Each wave owns one 16-column tile per iteration, reuses its B fragment
//   across the 4 row-tiles. Logits land in the soft_probs output region,
//   are re-read (L2-hot) for max / sum-exp, then overwritten with probs.
// ---------------------------------------------------------------------------
__global__ __launch_bounds__(256)
void k_gemm_softmax(const _Float16* __restrict__ xh,
                    const _Float16* __restrict__ cbh,
                    float* __restrict__ probs,
                    int*   __restrict__ idx_ws,
                    int*   __restrict__ counts,
                    float* __restrict__ idx_out) {
    __shared__ _Float16 As[ROWS_PER_BLK * D_DIM];   // 32 KB
    __shared__ float pmax[256];
    __shared__ int   pidx[256];
    __shared__ float rowm[ROWS_PER_BLK];
    __shared__ float rowsum[ROWS_PER_BLK];
    __shared__ int   rowi[ROWS_PER_BLK];

    const int t    = threadIdx.x;
    const int wave = t >> 5;
    const int lane = t & 31;
    const int half = lane >> 4;    // 0: lanes 0-15, 1: lanes 16-31
    const int ln   = lane & 15;
    const int rb   = blockIdx.x;   // row block (64 rows)

    // ---- stage A tile (64 x 256 f16, contiguous) into LDS --------------
    {
        const uint4* src = (const uint4*)(xh + (size_t)rb * ROWS_PER_BLK * D_DIM);
        uint4* dst = (uint4*)As;
        #pragma unroll
        for (int i = t; i < (ROWS_PER_BLK * D_DIM) / 8; i += 256) dst[i] = src[i];
    }
    __syncthreads();

    // ---- GEMM: 256 column tiles / 8 waves = 32 iterations --------------
    for (int it = 0; it < 32; ++it) {
        const int ct      = it * 8 + wave;
        const int colbase = ct * 16;

        v8f acc[4];
        #pragma unroll
        for (int rt = 0; rt < 4; ++rt)
            #pragma unroll
            for (int e = 0; e < 8; ++e) acc[rt][e] = 0.0f;

        // B fragment: lane = column (ln), 16 contiguous K starting at 16*half
        const _Float16* bbase = cbh + ((size_t)(colbase + ln)) * D_DIM + half * 16;

        #pragma unroll
        for (int ds = 0; ds < 8; ++ds) {           // D in steps of 32
            const v16h b = *(const v16h*)(bbase + ds * 32);
            #pragma unroll
            for (int rt = 0; rt < 4; ++rt) {
                // A fragment: lane = row (ln); K runs [16*half, +16) contiguous
                const _Float16* arow = As + (rt * 16 + ln) * D_DIM + ds * 32 + half * 16;
                union { v16h vec; v8h h[2]; } a;
                a.h[0] = *(const v8h*)(arow);
                a.h[1] = *(const v8h*)(arow + 8);
                acc[rt] = __builtin_amdgcn_wmma_f32_16x16x32_f16(
                    false, a.vec, false, b, (short)0, acc[rt], false, false);
            }
        }

        // store logits = 20 * cos.  C/D layout: VGPR r -> M = r + 8*half, lane -> N
        #pragma unroll
        for (int rt = 0; rt < 4; ++rt) {
            #pragma unroll
            for (int r = 0; r < 8; ++r) {
                const int grow = rb * ROWS_PER_BLK + rt * 16 + r + half * 8;
                probs[(size_t)grow * K_CB + colbase + ln] = LOGIT_SCALE * acc[rt][r];
            }
        }
    }

    __threadfence();
    __syncthreads();

    // ---- row max + argmax (4 threads per row, each scans 1024 cols) ----
    const int r = t >> 2;
    const int q = t & 3;
    const float* rp = probs + (size_t)(rb * ROWS_PER_BLK + r) * K_CB + q * 1024;

    float m = -1e30f; int mi = 0;
    for (int c = 0; c < 1024; ++c) {
        const float x = rp[c];
        if (x > m) { m = x; mi = q * 1024 + c; }   // strict > keeps first (argmin tie rule)
    }
    pmax[t] = m; pidx[t] = mi;
    __syncthreads();
    if (q == 0) {
        float bm = pmax[t]; int bi = pidx[t];
        #pragma unroll
        for (int j = 1; j < 4; ++j) {              // ascending order preserves ties
            if (pmax[t + j] > bm) { bm = pmax[t + j]; bi = pidx[t + j]; }
        }
        rowm[r] = bm; rowi[r] = bi;
    }
    __syncthreads();

    // ---- sum of exp(x - max) -------------------------------------------
    const float M = rowm[r];
    float s = 0.0f;
    for (int c = 0; c < 1024; ++c) s += __expf(rp[c] - M);
    pmax[t] = s;
    __syncthreads();
    if (q == 0) rowsum[r] = pmax[t] + pmax[t + 1] + pmax[t + 2] + pmax[t + 3];
    __syncthreads();

    // ---- hard indices + histogram --------------------------------------
    if (t < ROWS_PER_BLK) {
        const int grow = rb * ROWS_PER_BLK + t;
        const int id   = rowi[t];
        idx_ws[grow]  = id;
        idx_out[grow] = (float)id;
        atomicAdd(&counts[id], 1);
    }

    // ---- rewrite logits as probabilities (coalesced) --------------------
    for (int i = t; i < ROWS_PER_BLK * K_CB; i += 256) {
        const int rr = i >> 12;                    // i / 4096
        const size_t gi = (size_t)rb * ROWS_PER_BLK * K_CB + i;
        probs[gi] = __expf(probs[gi] - rowm[rr]) / rowsum[rr];
    }
}

// ---------------------------------------------------------------------------
// Gather quantized vectors (== straight-through values) + loss partials.
//   grid = 256 blocks x 256 threads; block handles 64 rows; thread = D lane.
// ---------------------------------------------------------------------------
__global__ __launch_bounds__(256)
void k_gather_loss(const float* __restrict__ cbn,
                   const float* __restrict__ xn,
                   const int*   __restrict__ idx_ws,
                   float* __restrict__ quant,
                   float* __restrict__ partials) {
    __shared__ float red[256];
    const int t = threadIdx.x;
    float acc = 0.0f;
    for (int j = 0; j < ROWS_PER_BLK; ++j) {
        const int grow = blockIdx.x * ROWS_PER_BLK + j;
        const int id   = idx_ws[grow];
        const float qv = cbn[(size_t)id * D_DIM + t];
        quant[(size_t)grow * D_DIM + t] = qv;
        const float d = qv - xn[(size_t)grow * D_DIM + t];
        acc += d * d;
    }
    red[t] = acc;
    __syncthreads();
    #pragma unroll
    for (int s = 128; s > 0; s >>= 1) {
        if (t < s) red[t] += red[t + s];
        __syncthreads();
    }
    if (t == 0) partials[blockIdx.x] = red[0];
}

// ---------------------------------------------------------------------------
// Deterministic finalize: vq_loss from partials, perplexity from histogram.
// ---------------------------------------------------------------------------
__global__ __launch_bounds__(256)
void k_finalize(const float* __restrict__ partials,
                const int*   __restrict__ counts,
                float* __restrict__ o_loss,
                float* __restrict__ o_perp) {
    __shared__ float red[256];
    const int t = threadIdx.x;

    red[t] = partials[t];                          // exactly 256 partials
    __syncthreads();
    #pragma unroll
    for (int s = 128; s > 0; s >>= 1) {
        if (t < s) red[t] += red[t + s];
        __syncthreads();
    }
    if (t == 0) o_loss[0] = red[0] * (0.25f / ((float)N_ROWS * (float)D_DIM));
    __syncthreads();

    float a = 0.0f;
    for (int k = t; k < K_CB; k += 256) {
        const float p = (float)counts[k] * (1.0f / (float)N_ROWS);
        a += p * logf(p + 1e-10f);
    }
    red[t] = a;
    __syncthreads();
    #pragma unroll
    for (int s = 128; s > 0; s >>= 1) {
        if (t < s) red[t] += red[t + s];
        __syncthreads();
    }
    if (t == 0) o_perp[0] = __expf(-red[0]);
}

__global__ void k_zero(int* __restrict__ counts) {
    for (int k = threadIdx.x; k < K_CB; k += 256) counts[k] = 0;
}

// ---------------------------------------------------------------------------
// Launch. Output layout (floats, concatenated in reference return order):
//   [0]                 vq_loss
//   [1 .. +4194304)     quant_st   (16,1024,256)
//   [.. +67108864)      soft_probs (16384,4096)
//   [71303169]          perplexity
//   [71303170 .. )      idx        (16384, cast to float)
// ---------------------------------------------------------------------------
extern "C" void kernel_launch(void* const* d_in, const int* in_sizes, int n_in,
                              void* d_out, int out_size, void* d_ws, size_t ws_size,
                              hipStream_t stream) {
    const float* inp = (const float*)d_in[0];   // [16,1024,256] f32
    const float* cb  = (const float*)d_in[1];   // [4096,256]    f32

    float* out     = (float*)d_out;
    float* o_loss  = out;
    float* o_quant = out + 1;
    float* o_probs = out + 1 + (size_t)N_ROWS * D_DIM;
    float* o_perp  = out + 1 + (size_t)N_ROWS * D_DIM + (size_t)N_ROWS * K_CB;
    float* o_idx   = o_perp + 1;

    char* w = (char*)d_ws;
    float*    xn      = (float*)(w);                       // 16 MB
    float*    cbn     = (float*)(w + 16777216);            //  4 MB
    _Float16* xh      = (_Float16*)(w + 20971520);         //  8 MB
    _Float16* cbh     = (_Float16*)(w + 29360128);         //  2 MB
    int*      idxw    = (int*)(w + 31457280);              // 64 KB
    int*      counts  = (int*)(w + 31522816);              // 16 KB
    float*    partial = (float*)(w + 31539200);            //  1 KB

    k_zero<<<1, 256, 0, stream>>>(counts);
    k_normalize<<<N_ROWS, 256, 0, stream>>>(inp, xn, xh);
    k_normalize<<<K_CB,   256, 0, stream>>>(cb,  cbn, cbh);
    k_gemm_softmax<<<N_ROWS / ROWS_PER_BLK, 256, 0, stream>>>(
        xh, cbh, o_probs, idxw, counts, o_idx);
    k_gather_loss<<<N_ROWS / ROWS_PER_BLK, 256, 0, stream>>>(
        cbn, xn, idxw, o_quant, partial);
    k_finalize<<<1, 256, 0, stream>>>(partial, counts, o_loss, o_perp);
}
